// SelfAttention_60224031424684
// MI455X (gfx1250) — matrix operand order repaired and checked by reading, not compiled, not run on hardware
//
#include <hip/hip_runtime.h>
#include <hip/hip_bf16.h>

typedef __bf16 bf16_t;
typedef __attribute__((ext_vector_type(8)))  __bf16 v8bf;
typedef __attribute__((ext_vector_type(16))) __bf16 v16bf;
typedef __attribute__((ext_vector_type(8)))  float  v8f;
typedef __attribute__((ext_vector_type(4)))  float  v4f;
typedef int v4i_vs __attribute__((vector_size(16)));   // matches builtin param type

#define DIM_    1024
#define SEQ_    2048
#define BATCH_  4
#define HEADS_  16
#define HD_     64
#define TOKENS_ (BATCH_*SEQ_)   // 8192

#if defined(__HIP_DEVICE_COMPILE__) && __has_builtin(__builtin_amdgcn_global_load_async_to_lds_b128)
#define HAS_ASYNC_LDS 1
#else
#define HAS_ASYNC_LDS 0
#endif

__device__ __forceinline__ bf16_t f2bf(float f) {
  unsigned u = __builtin_bit_cast(unsigned, f);
  unsigned short r = (unsigned short)((u + 0x7FFFu + ((u >> 16) & 1u)) >> 16);
  return __builtin_bit_cast(bf16_t, r);
}

__device__ __forceinline__ v8f wmma_bf16(v16bf a, v16bf b, v8f c) {
  return __builtin_amdgcn_wmma_f32_16x16x32_bf16(false, a, false, b, (short)0, c,
                                                 false, false);
}

// 16B global -> LDS copy: async (GLOBAL_LOAD_ASYNC_TO_LDS_B128, ASYNCcnt) when
// available, else synchronous VGPR round trip.
__device__ __forceinline__ void cp_async16(void* lds_dst, const void* g_src) {
#if HAS_ASYNC_LDS
  __builtin_amdgcn_global_load_async_to_lds_b128((v4i_vs*)g_src, (v4i_vs*)lds_dst,
                                                 0, 0);
#else
  *(v8bf*)lds_dst = *(const v8bf*)g_src;
#endif
}

__device__ __forceinline__ void wait_async0() {
#if HAS_ASYNC_LDS
#if __has_builtin(__builtin_amdgcn_s_wait_asynccnt)
  __builtin_amdgcn_s_wait_asynccnt(0);
#else
  asm volatile("s_wait_asynccnt 0" ::: "memory");
#endif
#endif
}

// A-operand fragment (16x32 bf16). Source row-major, ld in elements.
// Per ISA: lane half 0 holds K = {0..7, 16..23}, half 1 holds K = {8..15, 24..31}.
__device__ __forceinline__ v16bf load_a_frag(const bf16_t* p, int row, int ld,
                                             int k0, int half) {
  const bf16_t* r = p + (long)row * ld + k0;
  v8bf lo = *(const v8bf*)(r + half * 8);
  v8bf hi = *(const v8bf*)(r + 16 + half * 8);
  return __builtin_shufflevector(lo, hi, 0,1,2,3,4,5,6,7,8,9,10,11,12,13,14,15);
}

// B-operand fragment (32x16 bf16). Source is B^T row-major (N rows of K).
// Per ISA: lane half 0 holds K = {0..7, 16..23}, half 1 holds K = {8..15, 24..31}.
__device__ __forceinline__ v16bf load_b_frag(const bf16_t* p, int row, int ld,
                                             int k0, int half) {
  const bf16_t* r = p + (long)row * ld + k0;
  v8bf lo = *(const v8bf*)(r + half * 8);
  v8bf hi = *(const v8bf*)(r + 16 + half * 8);
  return __builtin_shufflevector(lo, hi, 0,1,2,3,4,5,6,7,8,9,10,11,12,13,14,15);
}

__device__ __forceinline__ float rowmax16(float v) {
#pragma unroll
  for (int m = 1; m < 16; m <<= 1) v = fmaxf(v, __shfl_xor(v, m, 32));
  return v;
}
__device__ __forceinline__ float rowsum16(float v) {
#pragma unroll
  for (int m = 1; m < 16; m <<= 1) v += __shfl_xor(v, m, 32);
  return v;
}

// ---------------------------------------------------------------------------
// fp32 -> bf16 cast, 8 elements / thread
// ---------------------------------------------------------------------------
__global__ void cast_f32_bf16(const float* __restrict__ in,
                              bf16_t* __restrict__ out, int n) {
  int i = (blockIdx.x * blockDim.x + threadIdx.x) * 8;
  if (i >= n) return;
  v4f a = *(const v4f*)(in + i);
  v4f b = *(const v4f*)(in + i + 4);
  v8bf o;
  o[0] = f2bf(a[0]); o[1] = f2bf(a[1]); o[2] = f2bf(a[2]); o[3] = f2bf(a[3]);
  o[4] = f2bf(b[0]); o[5] = f2bf(b[1]); o[6] = f2bf(b[2]); o[7] = f2bf(b[3]);
  *(v8bf*)(out + i) = o;
}

// ---------------------------------------------------------------------------
// WMMA GEMM: C[M,N] = A[M,K] * B^T[N,K]^T
// Block tile 128x128, 8 waves (4x2), wave tile 32x64.
// K staged 64-wide in LDS via async-to-LDS: 16 WMMAs per barrier pair.
// mode 0: QKV epilogue (scatter bf16 Q, K, V^T)   mode 1: fp32 + bias epilogue
// ---------------------------------------------------------------------------
__global__ __launch_bounds__(256)
void gemm_bf16(const bf16_t* __restrict__ A, const bf16_t* __restrict__ BT,
               int K, int mode,
               bf16_t* __restrict__ qb, bf16_t* __restrict__ kb,
               bf16_t* __restrict__ vtb,
               float* __restrict__ outF, const float* __restrict__ bias, int ldc) {
  __shared__ bf16_t Asm[128 * 64];
  __shared__ bf16_t Bsm[128 * 64];

  const int tid  = threadIdx.x;
  const int lane = tid & 31;
  const int wave = tid >> 5;
  const int half = lane >> 4;
  const int l16  = lane & 15;
  const int wm   = wave >> 1;   // 0..3 -> M offset *32
  const int wn   = wave & 1;    // 0..1 -> N offset *64
  const long rowBlock = (long)blockIdx.y * 128;
  const long colBlock = (long)blockIdx.x * 128;

  const v8f zero8 = {0.f, 0.f, 0.f, 0.f, 0.f, 0.f, 0.f, 0.f};
  v8f acc[2][4];
#pragma unroll
  for (int tm = 0; tm < 2; ++tm)
#pragma unroll
    for (int tn = 0; tn < 4; ++tn) acc[tm][tn] = zero8;

  const int ldr = tid >> 1;          // 0..127 tile row
  const int ldk = (tid & 1) * 32;    // k sub-offset 0/32
  const bf16_t* Arow = A  + (rowBlock + ldr) * K + ldk;
  const bf16_t* Brow = BT + (colBlock + ldr) * K + ldk;
  bf16_t* Alds = Asm + ldr * 64 + ldk;
  bf16_t* Blds = Bsm + ldr * 64 + ldk;

  for (int k0 = 0; k0 < K; k0 += 64) {
#pragma unroll
    for (int q = 0; q < 4; ++q) {
      cp_async16(Alds + q * 8, Arow + k0 + q * 8);
      cp_async16(Blds + q * 8, Brow + k0 + q * 8);
    }
    if (k0 + 64 < K) {                       // warm L2 for next slice
      __builtin_prefetch(Arow + k0 + 64, 0, 1);
      __builtin_prefetch(Brow + k0 + 64, 0, 1);
    }
    wait_async0();
    __syncthreads();

#pragma unroll
    for (int ks = 0; ks < 2; ++ks) {
      v16bf af[2], bfr[4];
#pragma unroll
      for (int tm = 0; tm < 2; ++tm)
        af[tm] = load_a_frag(Asm, wm * 32 + tm * 16 + l16, 64, ks * 32, half);
#pragma unroll
      for (int tn = 0; tn < 4; ++tn)
        bfr[tn] = load_b_frag(Bsm, wn * 64 + tn * 16 + l16, 64, ks * 32, half);
#pragma unroll
      for (int tm = 0; tm < 2; ++tm)
#pragma unroll
        for (int tn = 0; tn < 4; ++tn)
          acc[tm][tn] = wmma_bf16(af[tm], bfr[tn], acc[tm][tn]);
    }
    __syncthreads();
  }

#pragma unroll
  for (int tm = 0; tm < 2; ++tm) {
#pragma unroll
    for (int tn = 0; tn < 4; ++tn) {
      const long grow = rowBlock + wm * 32 + tm * 16 + half * 8;
      const int  gcol = (int)colBlock + wn * 64 + tn * 16 + l16;
      if (mode == 0) {
        const int which = gcol >> 10, fr = gcol & 1023;
        const int h = fr >> 6, hd = fr & 63;
#pragma unroll
        for (int j = 0; j < 8; ++j) {
          const long tok = grow + j;
          const int  b   = (int)(tok >> 11), n = (int)(tok & 2047);
          const long bh  = (long)b * HEADS_ + h;
          const bf16_t val = f2bf(acc[tm][tn][j]);
          if (which == 0)      qb [(bh * SEQ_ + n) * HD_ + hd]  = val;
          else if (which == 1) kb [(bh * SEQ_ + n) * HD_ + hd]  = val;
          else                 vtb[(bh * HD_ + hd) * SEQ_ + n]  = val;
        }
      } else {
        const float bv = bias[gcol];
#pragma unroll
        for (int j = 0; j < 8; ++j)
          outF[(grow + j) * ldc + gcol] = acc[tm][tn][j] + bv;
      }
    }
  }
}

// ---------------------------------------------------------------------------
// Flash attention. grid = (SEQ/128, BATCH*HEADS), block = 256 (8 waves).
// Each wave owns 16 query rows and full hd=64 fp32 accumulators.
// Key loop: 64 keys/iter; K tile + V^T tile async-staged in LDS by the block.
// ---------------------------------------------------------------------------
__global__ __launch_bounds__(256)
void attn_kernel(const bf16_t* __restrict__ qb, const bf16_t* __restrict__ kb,
                 const bf16_t* __restrict__ vtb, bf16_t* __restrict__ ob,
                 float scale) {
  __shared__ bf16_t Ksm[64 * 64];          // [key][hd]
  __shared__ bf16_t Vsm[64 * 64];          // [hd][key]  (V^T)
  __shared__ bf16_t Psm[8][16 * 64];       // per-wave P transpose staging

  const int tid  = threadIdx.x;
  const int lane = tid & 31;
  const int wave = tid >> 5;
  const int half = lane >> 4;
  const int l16  = lane & 15;
  const int bh   = blockIdx.y;                      // 0..63
  const int q0   = blockIdx.x * 128 + wave * 16;    // query row base (per batch)

  const bf16_t* Qbase = qb  + ((long)bh * SEQ_ + q0) * HD_;
  const bf16_t* Kbase = kb  + (long)bh * SEQ_ * HD_;
  const bf16_t* Vbase = vtb + (long)bh * HD_ * SEQ_;

  v16bf qf[2];
  qf[0] = load_a_frag(Qbase, l16, HD_, 0,  half);
  qf[1] = load_a_frag(Qbase, l16, HD_, 32, half);

  const v8f zero8 = {0.f, 0.f, 0.f, 0.f, 0.f, 0.f, 0.f, 0.f};
  v8f o[4];
#pragma unroll
  for (int t = 0; t < 4; ++t) o[t] = zero8;
  float mrow[8], lrow[8];
#pragma unroll
  for (int j = 0; j < 8; ++j) { mrow[j] = -3.0e38f; lrow[j] = 0.f; }

  const int r   = tid >> 2;          // 0..63
  const int c16 = (tid & 3) * 16;    // 0/16/32/48
  const bf16_t* Kld = Kbase + (long)r * HD_ + c16;
  const bf16_t* Vld = Vbase + (long)r * SEQ_ + c16;
  bf16_t* Klds = Ksm + r * 64 + c16;
  bf16_t* Vlds = Vsm + r * 64 + c16;

  for (int key0 = 0; key0 < SEQ_; key0 += 64) {
    cp_async16(Klds,     Kld + (long)key0 * HD_);
    cp_async16(Klds + 8, Kld + (long)key0 * HD_ + 8);
    cp_async16(Vlds,     Vld + key0);
    cp_async16(Vlds + 8, Vld + key0 + 8);
    if (key0 + 64 < SEQ_) {
      __builtin_prefetch(Kld + (long)(key0 + 64) * HD_, 0, 1);
      __builtin_prefetch(Vld + key0 + 64, 0, 1);
    }
    wait_async0();
    __syncthreads();

    // S = (Q K^T) * scale : 4 key sub-tiles x 2 k-steps
    v8f s[4];
#pragma unroll
    for (int t = 0; t < 4; ++t) {
      s[t] = zero8;
#pragma unroll
      for (int ks = 0; ks < 2; ++ks) {
        v16bf bfr = load_b_frag(Ksm, t * 16 + l16, 64, ks * 32, half);
        s[t] = wmma_bf16(qf[ks], bfr, s[t]);
      }
    }
#pragma unroll
    for (int t = 0; t < 4; ++t)
#pragma unroll
      for (int j = 0; j < 8; ++j) s[t][j] *= scale;

    // online softmax: rows live across 16-lane halves of one VGPR slot
#pragma unroll
    for (int j = 0; j < 8; ++j) {
      float rm = fmaxf(fmaxf(s[0][j], s[1][j]), fmaxf(s[2][j], s[3][j]));
      rm = rowmax16(rm);
      const float mn    = fmaxf(mrow[j], rm);
      const float alpha = __expf(mrow[j] - mn);
      mrow[j] = mn;
      float rs = 0.f;
#pragma unroll
      for (int t = 0; t < 4; ++t) {
        s[t][j] = __expf(s[t][j] - mn);
        rs += s[t][j];
      }
      rs = rowsum16(rs);
      lrow[j] = lrow[j] * alpha + rs;
#pragma unroll
      for (int t = 0; t < 4; ++t) o[t][j] *= alpha;
    }

    // transpose P (C layout -> A layout) through per-wave LDS
    bf16_t* P = Psm[wave];
#pragma unroll
    for (int t = 0; t < 4; ++t)
#pragma unroll
      for (int j = 0; j < 8; ++j)
        P[(half * 8 + j) * 64 + t * 16 + l16] = f2bf(s[t][j]);
    asm volatile("s_wait_dscnt 0" ::: "memory");

    // O += P V : 2 k-steps x 4 hd sub-tiles
#pragma unroll
    for (int ks = 0; ks < 2; ++ks) {
      v16bf pf = load_a_frag(P, l16, 64, ks * 32, half);
#pragma unroll
      for (int t = 0; t < 4; ++t) {
        v16bf vf = load_b_frag(Vsm, t * 16 + l16, 64, ks * 32, half);
        o[t] = wmma_bf16(pf, vf, o[t]);
      }
    }
    __syncthreads();
  }

  // normalize and store merged-head bf16 output [token][dim]
  const int b = bh >> 4, h = bh & 15;
#pragma unroll
  for (int t = 0; t < 4; ++t)
#pragma unroll
    for (int j = 0; j < 8; ++j) {
      const long tok = (long)b * SEQ_ + q0 + half * 8 + j;
      ob[tok * DIM_ + h * HD_ + t * 16 + l16] = f2bf(o[t][j] / lrow[j]);
    }
}

// ---------------------------------------------------------------------------
extern "C" void kernel_launch(void* const* d_in, const int* in_sizes, int n_in,
                              void* d_out, int out_size, void* d_ws, size_t ws_size,
                              hipStream_t stream) {
  const float* x     = (const float*)d_in[0];
  const float* w_qkv = (const float*)d_in[1];
  const float* w_out = (const float*)d_in[2];
  const float* b_out = (const float*)d_in[3];
  float* out = (float*)d_out;

  char* ws = (char*)d_ws;
  size_t off = 0;
  auto alloc = [&](size_t bytes) -> void* {
    void* p = ws + off;
    off += (bytes + 255) & ~(size_t)255;
    return p;
  };
  const size_t TD2 = (size_t)TOKENS_ * DIM_ * 2;
  bf16_t* xb  = (bf16_t*)alloc(TD2);                         // x bf16
  bf16_t* wqb = (bf16_t*)alloc((size_t)3 * DIM_ * DIM_ * 2); // w_qkv bf16
  bf16_t* wob = (bf16_t*)alloc((size_t)DIM_ * DIM_ * 2);     // w_out bf16
  bf16_t* qb  = (bf16_t*)alloc(TD2);                         // Q  [bh][n][hd]
  bf16_t* kb  = (bf16_t*)alloc(TD2);                         // K  [bh][n][hd]
  bf16_t* vtb = (bf16_t*)alloc(TD2);                         // V^T[bh][hd][n]
  bf16_t* ob  = (bf16_t*)alloc(TD2);                         // attn out bf16

  // 1) casts
  {
    int n = TOKENS_ * DIM_;
    cast_f32_bf16<<<n / (8 * 256), 256, 0, stream>>>(x, xb, n);
    n = 3 * DIM_ * DIM_;
    cast_f32_bf16<<<n / (8 * 256), 256, 0, stream>>>(w_qkv, wqb, n);
    n = DIM_ * DIM_;
    cast_f32_bf16<<<n / (8 * 256), 256, 0, stream>>>(w_out, wob, n);
  }

  // 2) QKV projection: [8192,1024] x [3072,1024]^T, scatter epilogue
  gemm_bf16<<<dim3(3 * DIM_ / 128, TOKENS_ / 128), 256, 0, stream>>>(
      xb, wqb, DIM_, 0, qb, kb, vtb, nullptr, nullptr, 0);

  // 3) flash attention; scale = DIM^-0.5 = 1/32 (full dim, per reference)
  attn_kernel<<<dim3(SEQ_ / 128, BATCH_ * HEADS_), 256, 0, stream>>>(
      qb, kb, vtb, ob, 0.03125f);

  // 4) out projection + bias: [8192,1024] x [1024,1024]^T -> fp32
  gemm_bf16<<<dim3(DIM_ / 128, TOKENS_ / 128), 256, 0, stream>>>(
      ob, wob, DIM_, 1, nullptr, nullptr, nullptr, out, b_out, DIM_);
}
